// SPELL_65807488909454
// MI455X (gfx1250) — compile-verified, matrix operand order checked by reading, not run on hardware
//
#include <hip/hip_runtime.h>
#include <hip/hip_bf16.h>

// ---------------------------------------------------------------------------
// Problem constants (from the reference)
// ---------------------------------------------------------------------------
#define NN   50000
#define EE   1000000
#define FD_  1024
#define XW   1028        // x row stride (FD + 4)

typedef __attribute__((ext_vector_type(16))) _Float16 v16h;
typedef __attribute__((ext_vector_type(8)))  _Float16 v8h;
typedef __attribute__((ext_vector_type(8)))  float    v8f;
typedef int v4i __attribute__((vector_size(16)));

#define AS1 __attribute__((address_space(1)))
#define AS3 __attribute__((address_space(3)))

#if defined(__has_builtin)
#if __has_builtin(__builtin_amdgcn_global_load_async_to_lds_b128)
#define HAVE_ASYNC_LDS 1
#endif
#if __has_builtin(__builtin_amdgcn_s_wait_asynccnt)
#define HAVE_WAIT_ASYNC_BUILTIN 1
#endif
#endif

// ---------------------------------------------------------------------------
// Helpers
// ---------------------------------------------------------------------------
__device__ inline v16h mk16(v8h a, v8h b) {
    return __builtin_shufflevector(a, b, 0,1,2,3,4,5,6,7,8,9,10,11,12,13,14,15);
}

__device__ inline v8f wmma_f16(v16h a, v16h b, v8f c) {
    // v_wmma_f32_16x16x32_f16
    return __builtin_amdgcn_wmma_f32_16x16x32_f16(false, a, false, b,
                                                  (short)0, c, false, false);
}

__device__ inline v8h ld8h(const _Float16* p) { return *(const v8h*)p; }

__device__ inline v8h cvt8(const float* p) {
    const float4* q = (const float4*)p;
    float4 a = q[0], b = q[1];
    v8h r;
    r[0]=(_Float16)a.x; r[1]=(_Float16)a.y; r[2]=(_Float16)a.z; r[3]=(_Float16)a.w;
    r[4]=(_Float16)b.x; r[5]=(_Float16)b.y; r[6]=(_Float16)b.z; r[7]=(_Float16)b.w;
    return r;
}

__device__ inline v8h cvt8s(const float* p, float sc) {
    const float4* q = (const float4*)p;
    float4 a = q[0], b = q[1];
    v8h r;
    r[0]=(_Float16)(a.x*sc); r[1]=(_Float16)(a.y*sc); r[2]=(_Float16)(a.z*sc); r[3]=(_Float16)(a.w*sc);
    r[4]=(_Float16)(b.x*sc); r[5]=(_Float16)(b.y*sc); r[6]=(_Float16)(b.z*sc); r[7]=(_Float16)(b.w*sc);
    return r;
}

// B fragment loader (fragment-ordered buffer; works for global or LDS pointers)
__device__ inline v16h loadB(const _Float16* Wp, int kc, int nt, int lane) {
    const _Float16* p = Wp + (((size_t)(kc*4 + nt)*32 + lane) * 16);
    return *(const v16h*)p;
}

// float atomic max via signed/unsigned int ordering trick
__device__ inline void atomicMaxF(float* a, float v) {
    if (v >= 0.0f) atomicMax((int*)a, __float_as_int(v));
    else           atomicMin((unsigned int*)a, __float_as_uint(v));
}

// cooperative global->LDS copy of 16B chunks (async path when available)
__device__ inline void stage16(const _Float16* g, _Float16* l) {
#ifdef HAVE_ASYNC_LDS
    __builtin_amdgcn_global_load_async_to_lds_b128(
        (AS1 v4i*)g, (AS3 v4i*)l, 0, 0);
#else
    *(v8h*)l = *(const v8h*)g;
#endif
}

__device__ inline void wait_async_lds() {
#ifdef HAVE_ASYNC_LDS
#ifdef HAVE_WAIT_ASYNC_BUILTIN
    __builtin_amdgcn_s_wait_asynccnt(0);
#else
    asm volatile("s_wait_asynccnt 0x0" ::: "memory");
#endif
#endif
}

// ---------------------------------------------------------------------------
// Weight packing: W (nOut x K, row-major f32) -> f16 WMMA B-fragments.
// ---------------------------------------------------------------------------
__global__ void pack_b_kernel(const float* __restrict__ W, _Float16* __restrict__ dst,
                              int K, int nOut) {
    int idx = blockIdx.x * blockDim.x + threadIdx.x;
    int total = (K/32) * (nOut/16) * 512;
    if (idx >= total) return;
    int j    = idx & 15;
    int lane = (idx >> 4) & 31;
    int frag = idx >> 9;
    int ntiles = nOut / 16;
    int kc = frag / ntiles;
    int nt = frag % ntiles;
    int n  = nt*16 + (lane & 15);
    int kh = lane >> 4;
    int k  = kc*32 + ((j < 8) ? (kh*8 + j) : (16 + kh*8 + (j - 8)));
    dst[idx] = (_Float16)W[(size_t)n * K + k];
}

// ---------------------------------------------------------------------------
// Stage A: h = relu(bn01(x_aud + x_vis)). One wave per 16-node tile.
// ---------------------------------------------------------------------------
__global__ __launch_bounds__(256)
void node_proj_kernel(const float* __restrict__ x,
                      const float* __restrict__ Wspf, const float* __restrict__ bspf,
                      const _Float16* __restrict__ W011p, const _Float16* __restrict__ W012p,
                      const float* __restrict__ b011, const float* __restrict__ b012,
                      const float* __restrict__ bn01,
                      float* __restrict__ h32, _Float16* __restrict__ h16, int nTiles) {
    __shared__ __align__(32) _Float16 spfS[8][16*64];
    int wave = threadIdx.x >> 5, lane = threadIdx.x & 31;
    int tile = blockIdx.x * 8 + wave;
    if (tile >= nTiles) return;
    int row0 = tile * 16;

    // spf projection (16x4 @ 4x64 + bias) into LDS as f16
    _Float16* sp = spfS[wave];
    for (int t = 0; t < 32; t++) {
        int idx = lane * 32 + t;
        int m = idx >> 6, n = idx & 63;
        const float* xr = x + (size_t)(row0 + m) * XW + FD_;
        float v = bspf[n];
        #pragma unroll
        for (int q = 0; q < 4; q++) v += xr[q] * Wspf[n*4 + q];
        sp[m*64 + n] = (_Float16)v;
    }
    asm volatile("s_wait_dscnt 0" ::: "memory");

    int m = lane & 15, kh = lane >> 4;
    const float* xrow = x + (size_t)(row0 + m) * XW;
    v8f zero = {0,0,0,0,0,0,0,0};
    v8f acc[4] = {zero, zero, zero, zero};

    for (int kc = 0; kc < 16; kc++) {           // vis x-part (cols 512..1023)
        const float* base = xrow + 512 + kc*32;
        v16h a = mk16(cvt8(base + kh*8), cvt8(base + 16 + kh*8));
        #pragma unroll
        for (int nt = 0; nt < 4; nt++)
            acc[nt] = wmma_f16(a, loadB(W011p, kc, nt, lane), acc[nt]);
    }
    for (int kc = 0; kc < 2; kc++) {            // vis spf-part (LDS)
        const _Float16* base = sp + m*64 + kc*32;
        v16h a = mk16(ld8h(base + kh*8), ld8h(base + 16 + kh*8));
        #pragma unroll
        for (int nt = 0; nt < 4; nt++)
            acc[nt] = wmma_f16(a, loadB(W011p, 16 + kc, nt, lane), acc[nt]);
    }
    for (int kc = 0; kc < 16; kc++) {           // aud (cols 0..511)
        const float* base = xrow + kc*32;
        v16h a = mk16(cvt8(base + kh*8), cvt8(base + 16 + kh*8));
        #pragma unroll
        for (int nt = 0; nt < 4; nt++)
            acc[nt] = wmma_f16(a, loadB(W012p, kc, nt, lane), acc[nt]);
    }

    #pragma unroll
    for (int nt = 0; nt < 4; nt++) {
        int n = (lane & 15) + nt*16;
        float g = bn01[n], bb = bn01[64+n], mm = bn01[128+n], vv = bn01[192+n];
        float sc = rsqrtf(vv + 1e-5f) * g;
        float bias = b011[n] + b012[n];
        #pragma unroll
        for (int r = 0; r < 8; r++) {
            int mr = r + 8*kh;
            float v = acc[nt][r] + bias;
            v = (v - mm) * sc + bb;
            v = fmaxf(v, 0.0f);
            size_t o = (size_t)(row0 + mr) * 64 + n;
            h32[o] = v; h16[o] = (_Float16)v;
        }
    }
}

// ---------------------------------------------------------------------------
// Stage B: EdgeConv. Weights staged into LDS once per block (async-to-LDS),
// one wave per 16-edge tile; GEMM1 (K=128) -> relu -> LDS -> GEMM2 (K=64)
// -> atomic segment-max.
// ---------------------------------------------------------------------------
__global__ __launch_bounds__(256)
void edge_conv_kernel(const int* __restrict__ ei, const int* __restrict__ ea,
                      const _Float16* __restrict__ h16,
                      const _Float16* __restrict__ W1p, const float* __restrict__ b1,
                      const _Float16* __restrict__ W2p, const float* __restrict__ b2,
                      float* __restrict__ eo, int branch, int nTiles) {
    __shared__ __align__(32) _Float16 m1S[8][16*64];
    __shared__ __align__(32) _Float16 w1S[16*512];   // 16 fragments = 16 KB
    __shared__ __align__(32) _Float16 w2S[8*512];    //  8 fragments =  8 KB
    __shared__ int segS[8][16];

    // --- cooperative weight staging (all 256 threads) ---
    {
        int tid = threadIdx.x;
        #pragma unroll
        for (int r = 0; r < 4; r++) {                // W1: 1024 x 16B chunks
            int off = (tid + r*256) * 8;
            stage16(W1p + off, w1S + off);
        }
        #pragma unroll
        for (int r = 0; r < 2; r++) {                // W2: 512 x 16B chunks
            int off = (tid + r*256) * 8;
            stage16(W2p + off, w2S + off);
        }
        wait_async_lds();
    }
    __syncthreads();

    int wave = threadIdx.x >> 5, lane = threadIdx.x & 31;
    int tile = blockIdx.x * 8 + wave;
    if (tile >= nTiles) return;
    int m = lane & 15, kh = lane >> 4;
    int e = tile * 16 + m;
    int srcI = ei[e], dstI = ei[EE + e];
    if (lane < 16) {
        int a = ea[e];
        bool keep = (branch == 0) ? (a >= 0) : ((branch == 1) ? (a <= 0) : true);
        segS[wave][lane] = keep ? dstI : -1;
    }
    const _Float16* hs = h16 + (size_t)srcI * 64;
    const _Float16* hd = h16 + (size_t)dstI * 64;

    v8f zero = {0,0,0,0,0,0,0,0};
    v8f acc[4] = {zero, zero, zero, zero};
    #pragma unroll
    for (int kc = 0; kc < 4; kc++) {
        v16h a;
        if (kc < 2) {                                // x_i block
            const _Float16* b_ = hd + kc*32;
            a = mk16(ld8h(b_ + kh*8), ld8h(b_ + 16 + kh*8));
        } else {                                     // (x_j - x_i) block
            int c = (kc - 2) * 32;
            v8h lo = ld8h(hs + c + kh*8)      - ld8h(hd + c + kh*8);
            v8h hi = ld8h(hs + c + 16 + kh*8) - ld8h(hd + c + 16 + kh*8);
            a = mk16(lo, hi);
        }
        #pragma unroll
        for (int nt = 0; nt < 4; nt++)
            acc[nt] = wmma_f16(a, loadB(w1S, kc, nt, lane), acc[nt]);
    }

    // relu(. + b1) -> LDS tile (row-major 16x64 f16) for A-layout of GEMM2
    _Float16* mp = m1S[wave];
    #pragma unroll
    for (int nt = 0; nt < 4; nt++) {
        int n = (lane & 15) + nt*16;
        float bv = b1[n];
        #pragma unroll
        for (int r = 0; r < 8; r++) {
            int mr = r + 8*kh;
            mp[mr*64 + n] = (_Float16)fmaxf(acc[nt][r] + bv, 0.0f);
        }
    }
    asm volatile("s_wait_dscnt 0" ::: "memory");

    v8f acc2[4] = {zero, zero, zero, zero};
    #pragma unroll
    for (int kc = 0; kc < 2; kc++) {
        const _Float16* base = mp + m*64 + kc*32;
        v16h a = mk16(ld8h(base + kh*8), ld8h(base + 16 + kh*8));
        #pragma unroll
        for (int nt = 0; nt < 4; nt++)
            acc2[nt] = wmma_f16(a, loadB(w2S, kc, nt, lane), acc2[nt]);
    }

    #pragma unroll
    for (int nt = 0; nt < 4; nt++) {
        int n = (lane & 15) + nt*16;
        float bv = b2[n];
        #pragma unroll
        for (int r = 0; r < 8; r++) {
            int mr = r + 8*kh;
            int seg = segS[wave][mr];
            if (seg >= 0) atomicMaxF(eo + (size_t)seg * 64 + n, acc2[nt][r] + bv);
        }
    }
}

// ---------------------------------------------------------------------------
// Elementwise: y = relu(bn(where(isfinite(eo), eo, 0))); write f32 + f16.
// ---------------------------------------------------------------------------
__global__ void bnrelu_kernel(const float* __restrict__ eo, const float* __restrict__ bn,
                              float* __restrict__ y32, _Float16* __restrict__ y16, size_t n) {
    size_t i = (size_t)blockIdx.x * blockDim.x + threadIdx.x;
    if (i >= n) return;
    int c = (int)(i & 63);
    float v = eo[i];
    if (!(v > -INFINITY && v < INFINITY)) v = 0.0f;   // NaN / +-inf -> 0
    v = (v - bn[128 + c]) * rsqrtf(bn[192 + c] + 1e-5f) * bn[c] + bn[64 + c];
    v = fmaxf(v, 0.0f);
    y32[i] = v; y16[i] = (_Float16)v;
}

// ---------------------------------------------------------------------------
// Edge scatter-add for SAGE: one WAVE per edge. Scalar index loads, coalesced
// float2-per-lane row read, 64 atomic adds spread across 32 lanes.
// ---------------------------------------------------------------------------
__global__ __launch_bounds__(256)
void scatter_sum_kernel(const int* __restrict__ ei, const int* __restrict__ ea,
                        const float* __restrict__ feat,
                        float* __restrict__ s, float* __restrict__ cnt, int branch) {
    int t = blockIdx.x * blockDim.x + threadIdx.x;
    int e = t >> 5, lane = t & 31;
    if (e >= EE) return;
    int a = ea[e];
    bool keep = (branch == 0) ? (a >= 0) : ((branch == 1) ? (a <= 0) : true);
    if (!keep) return;
    int src = ei[e], dst = ei[EE + e];
    float2 v = ((const float2*)(feat + (size_t)src * 64))[lane];
    float* sp = s + (size_t)dst * 64 + lane * 2;
    atomicAdd(sp,     v.x);
    atomicAdd(sp + 1, v.y);
    if (lane == 0) atomicAdd(&cnt[dst], 1.0f);
}

// ---------------------------------------------------------------------------
// SAGE node GEMM: y2 = relu(bn21( (s/max(cnt,1)) @ Wl.T + bl + y @ Wr.T )).
// ---------------------------------------------------------------------------
__global__ __launch_bounds__(256)
void sage_node_kernel(const float* __restrict__ s, const float* __restrict__ cnt,
                      const _Float16* __restrict__ yin16,
                      const _Float16* __restrict__ Wlp, const _Float16* __restrict__ Wrp,
                      const float* __restrict__ bl, const float* __restrict__ bn,
                      float* __restrict__ yo32, _Float16* __restrict__ yo16, int nTiles) {
    int wave = threadIdx.x >> 5, lane = threadIdx.x & 31;
    int tile = blockIdx.x * 8 + wave;
    if (tile >= nTiles) return;
    int m = lane & 15, kh = lane >> 4;
    int node = tile * 16 + m;
    float inv = 1.0f / fmaxf(cnt[node], 1.0f);
    const float*    srow = s     + (size_t)node * 64;
    const _Float16* yrow = yin16 + (size_t)node * 64;

    v8f zero = {0,0,0,0,0,0,0,0};
    v8f acc[4] = {zero, zero, zero, zero};
    #pragma unroll
    for (int kc = 0; kc < 2; kc++) {
        v16h aa = mk16(cvt8s(srow + kc*32 + kh*8, inv),
                       cvt8s(srow + kc*32 + 16 + kh*8, inv));
        #pragma unroll
        for (int nt = 0; nt < 4; nt++)
            acc[nt] = wmma_f16(aa, loadB(Wlp, kc, nt, lane), acc[nt]);
        v16h ay = mk16(ld8h(yrow + kc*32 + kh*8), ld8h(yrow + kc*32 + 16 + kh*8));
        #pragma unroll
        for (int nt = 0; nt < 4; nt++)
            acc[nt] = wmma_f16(ay, loadB(Wrp, kc, nt, lane), acc[nt]);
    }

    int row0 = tile * 16;
    #pragma unroll
    for (int nt = 0; nt < 4; nt++) {
        int n = (lane & 15) + nt*16;
        float g = bn[n], bb = bn[64+n], mm = bn[128+n], vv = bn[192+n];
        float sc = rsqrtf(vv + 1e-5f) * g;
        float bias = bl[n];
        #pragma unroll
        for (int r = 0; r < 8; r++) {
            int mr = r + 8*kh;
            float v = acc[nt][r] + bias;
            v = (v - mm) * sc + bb;
            v = fmaxf(v, 0.0f);
            size_t o = (size_t)(row0 + mr) * 64 + n;
            yo32[o] = v; yo16[o] = (_Float16)v;
        }
    }
}

// ---------------------------------------------------------------------------
// Final SAGE (width 1): one WAVE per node, float2 lanes + shuffle reduction.
// ---------------------------------------------------------------------------
__global__ __launch_bounds__(256)
void sage_out_kernel(const float* __restrict__ s, const float* __restrict__ cnt,
                     const float* __restrict__ y,
                     const float* __restrict__ Wl, const float* __restrict__ bl,
                     const float* __restrict__ Wr, float* __restrict__ o) {
    int t = blockIdx.x * blockDim.x + threadIdx.x;
    int i = t >> 5, lane = t & 31;
    if (i >= NN) return;
    float inv = 1.0f / fmaxf(cnt[i], 1.0f);
    float2 sv = ((const float2*)(s + (size_t)i * 64))[lane];
    float2 yv = ((const float2*)(y + (size_t)i * 64))[lane];
    float2 wl = ((const float2*)Wl)[lane];
    float2 wr = ((const float2*)Wr)[lane];
    float acc = (sv.x * inv) * wl.x + (sv.y * inv) * wl.y + yv.x * wr.x + yv.y * wr.y;
    #pragma unroll
    for (int off = 16; off > 0; off >>= 1) acc += __shfl_xor(acc, off, 32);
    if (lane == 0) o[i] += acc + bl[0];
}

__global__ void fill_kernel(float* __restrict__ p, size_t n, float v) {
    size_t i = (size_t)blockIdx.x * blockDim.x + threadIdx.x;
    if (i < n) p[i] = v;
}

__global__ void sigmoid_kernel(const float* __restrict__ o, float* __restrict__ out) {
    int i = blockIdx.x * blockDim.x + threadIdx.x;
    if (i >= NN) return;
    out[i] = 1.0f / (1.0f + expf(-o[i]));
}

// ---------------------------------------------------------------------------
// Host orchestration
// ---------------------------------------------------------------------------
extern "C" void kernel_launch(void* const* d_in, const int* in_sizes, int n_in,
                              void* d_out, int out_size, void* d_ws, size_t ws_size,
                              hipStream_t stream) {
    const float* x    = (const float*)d_in[0];
    const int*   ei   = (const int*)  d_in[1];
    const int*   ea   = (const int*)  d_in[2];
    const float* Wspf = (const float*)d_in[3];
    const float* bspf = (const float*)d_in[4];
    const float* W011 = (const float*)d_in[5];
    const float* b011 = (const float*)d_in[6];
    const float* W012 = (const float*)d_in[7];
    const float* b012 = (const float*)d_in[8];
    const float* bn01 = (const float*)d_in[9];
    const float* eW1[3] = {(const float*)d_in[10], (const float*)d_in[15], (const float*)d_in[20]};
    const float* eb1[3] = {(const float*)d_in[11], (const float*)d_in[16], (const float*)d_in[21]};
    const float* eW2[3] = {(const float*)d_in[12], (const float*)d_in[17], (const float*)d_in[22]};
    const float* eb2[3] = {(const float*)d_in[13], (const float*)d_in[18], (const float*)d_in[23]};
    const float* bnE[3] = {(const float*)d_in[14], (const float*)d_in[19], (const float*)d_in[24]};
    const float* s21Wl = (const float*)d_in[25];
    const float* s21bl = (const float*)d_in[26];
    const float* s21Wr = (const float*)d_in[27];
    const float* bn21  = (const float*)d_in[28];
    const float* s3Wl[3] = {(const float*)d_in[29], (const float*)d_in[32], (const float*)d_in[35]};
    const float* s3bl[3] = {(const float*)d_in[30], (const float*)d_in[33], (const float*)d_in[36]};
    const float* s3Wr[3] = {(const float*)d_in[31], (const float*)d_in[34], (const float*)d_in[37]};

    // --- carve workspace ---
    char* wp = (char*)d_ws;
    auto carve = [&](size_t bytes) {
        void* p = (void*)wp;
        wp += (bytes + 255) & ~(size_t)255;
        return p;
    };
    const size_t NF = (size_t)NN * 64;
    float*    h32   = (float*)   carve(NF * 4);
    _Float16* h16   = (_Float16*)carve(NF * 2);
    float*    y32   = (float*)   carve(NF * 4);
    _Float16* y16   = (_Float16*)carve(NF * 2);
    float*    y232  = (float*)   carve(NF * 4);
    _Float16* y216  = (_Float16*)carve(NF * 2);
    float*    eo    = (float*)   carve(NF * 4);   // edge-max buffer, aliased as ssum
    float*    cnt   = (float*)   carve((size_t)NN * 4);
    float*    oacc  = (float*)   carve((size_t)NN * 4);
    _Float16* W011p = (_Float16*)carve((size_t)18*4*512 * 2);
    _Float16* W012p = (_Float16*)carve((size_t)16*4*512 * 2);
    _Float16* eW1p[3], *eW2p[3];
    for (int b = 0; b < 3; b++) {
        eW1p[b] = (_Float16*)carve((size_t)4*4*512 * 2);
        eW2p[b] = (_Float16*)carve((size_t)2*4*512 * 2);
    }
    _Float16* s21Wlp = (_Float16*)carve((size_t)2*4*512 * 2);
    _Float16* s21Wrp = (_Float16*)carve((size_t)2*4*512 * 2);
    float* ssum = eo;   // eo is dead after bnrelu_kernel consumes it

    auto pack = [&](const float* W, _Float16* dst, int K, int nOut) {
        int total = (K/32) * (nOut/16) * 512;
        pack_b_kernel<<<(total + 255)/256, 256, 0, stream>>>(W, dst, K, nOut);
    };
    pack(W011, W011p, 576, 64);
    pack(W012, W012p, 512, 64);
    for (int b = 0; b < 3; b++) {
        pack(eW1[b], eW1p[b], 128, 64);
        pack(eW2[b], eW2p[b],  64, 64);
    }
    pack(s21Wl, s21Wlp, 64, 64);
    pack(s21Wr, s21Wrp, 64, 64);

    const int nodeTiles = NN / 16;                 // 3125
    node_proj_kernel<<<(nodeTiles + 7)/8, 256, 0, stream>>>(
        x, Wspf, bspf, W011p, W012p, b011, b012, bn01, h32, h16, nodeTiles);

    fill_kernel<<<(int)((NN + 255)/256), 256, 0, stream>>>(oacc, (size_t)NN, 0.0f);

    const int edgeTiles = EE / 16;                 // 62500
    const int scatterBlocks = (int)(((size_t)EE * 32 + 255) / 256);
    const int waveNodeBlocks = (int)(((size_t)NN * 32 + 255) / 256);
    for (int br = 0; br < 3; br++) {
        fill_kernel<<<(int)((NF + 255)/256), 256, 0, stream>>>(eo, NF, -INFINITY);
        edge_conv_kernel<<<(edgeTiles + 7)/8, 256, 0, stream>>>(
            ei, ea, h16, eW1p[br], eb1[br], eW2p[br], eb2[br], eo, br, edgeTiles);
        bnrelu_kernel<<<(int)((NF + 255)/256), 256, 0, stream>>>(eo, bnE[br], y32, y16, NF);

        fill_kernel<<<(int)((NF + 255)/256), 256, 0, stream>>>(ssum, NF, 0.0f);
        fill_kernel<<<(int)((NN + 255)/256), 256, 0, stream>>>(cnt, (size_t)NN, 0.0f);
        scatter_sum_kernel<<<scatterBlocks, 256, 0, stream>>>(ei, ea, y32, ssum, cnt, br);
        sage_node_kernel<<<(nodeTiles + 7)/8, 256, 0, stream>>>(
            ssum, cnt, y16, s21Wlp, s21Wrp, s21bl, bn21, y232, y216, nodeTiles);

        fill_kernel<<<(int)((NF + 255)/256), 256, 0, stream>>>(ssum, NF, 0.0f);
        fill_kernel<<<(int)((NN + 255)/256), 256, 0, stream>>>(cnt, (size_t)NN, 0.0f);
        scatter_sum_kernel<<<scatterBlocks, 256, 0, stream>>>(ei, ea, y232, ssum, cnt, br);
        sage_out_kernel<<<waveNodeBlocks, 256, 0, stream>>>(
            ssum, cnt, y232, s3Wl[br], s3bl[br], s3Wr[br], oacc);
    }

    sigmoid_kernel<<<(NN + 255)/256, 256, 0, stream>>>(oacc, (float*)d_out);
    (void)in_sizes; (void)n_in; (void)out_size; (void)ws_size;
}
